// _ScaledDotProductAttention_Weighted_7868380086834
// MI455X (gfx1250) — matrix operand order repaired and checked by reading, not compiled, not run on hardware
//
#include <hip/hip_runtime.h>
#include <hip/hip_bf16.h>

typedef __attribute__((ext_vector_type(16))) _Float16 v16h;
typedef __attribute__((ext_vector_type(8)))  float    v8f;

#define BSZ     4
#define NH      16
#define BH      64          // BSZ*NH
#define QL      1024
#define SL      1024
#define HD      64
#define MTILE   32          // q rows per workgroup
#define WSTRIDE 1048        // f16 LDS row stride: bank-conflict-free row gather, 16B-aligned rows

// d_out layout (reference return order: output, attn_weights, attn_scores)
#define W_OFF   ((size_t)BH * QL * HD)                    // 4,194,304
#define S_OFF   (W_OFF + (size_t)BH * QL * SL)            // 71,303,168

// dynamic LDS partition (total 104,448 B -> 3 workgroups per 320KB WGP)
#define LDS_W_BYTES   (MTILE * WSTRIDE * 2)               // 67,072 (f16 scores/weights)
#define LDS_Q_BYTES   (MTILE * 72 * 2)                    //  4,608 (stride-72 f16)
#define LDS_KV_BYTES  (64 * 64 * 4)                       // 16,384 per staging buffer (f32)
#define LDS_TOTAL     (LDS_W_BYTES + LDS_Q_BYTES + 2 * LDS_KV_BYTES)

// ---- CDNA5 async global->LDS copy (ASYNCcnt-tracked, bypasses VGPRs) ----
__device__ __forceinline__ void async_load_b128(void* lds_ptr, const void* gptr) {
    unsigned int lds_off = (unsigned int)(uintptr_t)lds_ptr;   // flat->LDS low-32 truncation (ISA 10.2)
    unsigned long long ga = (unsigned long long)(uintptr_t)gptr;
    asm volatile("global_load_async_to_lds_b128 %0, %1, off"
                 :: "v"(lds_off), "v"(ga)
                 : "memory");
}

__device__ __forceinline__ void wait_async0() {
#if __has_builtin(__builtin_amdgcn_s_wait_asynccnt)
    __builtin_amdgcn_s_wait_asynccnt(0);
#else
    asm volatile("s_wait_asynccnt 0" ::: "memory");
#endif
}

__global__ __launch_bounds__(256)
void attn_fused_kernel(const float* __restrict__ q,
                       const float* __restrict__ k,
                       const float* __restrict__ v,
                       const float* __restrict__ prev,
                       const float* __restrict__ mask,
                       const float* __restrict__ scale_p,
                       float* __restrict__ out)
{
    extern __shared__ char smem[];
    _Float16* s_w   = (_Float16*)smem;                                   // [32][1048] f16 scores->weights
    _Float16* s_q   = (_Float16*)(smem + LDS_W_BYTES);                   // [32][72]   f16
    float*    s_kv0 = (float*)(smem + LDS_W_BYTES + LDS_Q_BYTES);        // [64][64]   f32 staging A
    float*    s_kv1 = (float*)(smem + LDS_W_BYTES + LDS_Q_BYTES + LDS_KV_BYTES); // staging B

    const int bh  = blockIdx.x >> 5;          // 0..63
    const int m0  = (blockIdx.x & 31) * MTILE;
    const int h   = bh & (NH - 1);
    const int tid = threadIdx.x;
    const int wave = tid >> 5;
    const int lane = tid & 31;
    const int hs   = lane >> 4;               // half-wave select
    const int n    = lane & 15;               // N / row-within-half
    const int mi   = wave >> 2;               // 0..1 : 16-row half of the 32-row tile
    const int si   = wave & 3;                // 0..3 : 16-col sub-tile

    const float scl = scale_p[0];
    const size_t kbase = (size_t)bh * HD * SL;   // k is [BH][64][1024]

    // async-issue helpers (per-thread slices of the cooperative tile copies)
    auto issue_k = [&](int sc, float* buf) {
        int d  = tid >> 2;                    // 0..63
        int c0 = (tid & 3) * 16;              // 16 floats -> 4x b128
        const float* gsrc = k + kbase + (size_t)d * SL + sc * 64 + c0;
        float* ldst = buf + d * 64 + c0;
#pragma unroll
        for (int u = 0; u < 4; ++u)
            async_load_b128(ldst + u * 4, gsrc + u * 4);
    };
    auto issue_v = [&](int kc, float* buf) {
        int r  = tid >> 3;                    // 0..31
        int c0 = (tid & 7) * 8;               // 8 floats -> 2x b128
        const float* gsrc = v + ((size_t)(bh * SL + kc * 32 + r)) * HD + c0;
        float* ldst = buf + r * 64 + c0;
        async_load_b128(ldst,     gsrc);
        async_load_b128(ldst + 4, gsrc + 4);
    };

    // ---- Stage 1: q tile [32x64] f32 -> f16 LDS (read once, keep converted) ----
    {
        int r  = tid >> 3;
        int c0 = (tid & 7) * 8;
        const float* src = q + ((size_t)(bh * QL + m0 + r)) * HD + c0;
        float4 f0 = *(const float4*)(src);
        float4 f1 = *(const float4*)(src + 4);
        _Float16* dst = s_q + r * 72 + c0;
        dst[0] = (_Float16)f0.x; dst[1] = (_Float16)f0.y;
        dst[2] = (_Float16)f0.z; dst[3] = (_Float16)f0.w;
        dst[4] = (_Float16)f1.x; dst[5] = (_Float16)f1.y;
        dst[6] = (_Float16)f1.z; dst[7] = (_Float16)f1.w;
    }

    // ---- Stage 2: scores = q@k * scale + prev ; double-buffered async k stream ----
    issue_k(0, s_kv0);
    for (int sc = 0; sc < SL / 64; ++sc) {
        float* cur = (sc & 1) ? s_kv1 : s_kv0;
        float* nxt = (sc & 1) ? s_kv0 : s_kv1;
        wait_async0();                 // my chunk-sc copies landed
        __syncthreads();               // everyone's landed; prior compute done with 'nxt'
        if (sc + 1 < SL / 64) issue_k(sc + 1, nxt);   // overlap next copy with this compute

        v8f acc = {};
#pragma unroll
        for (int kk = 0; kk < 2; ++kk) {          // K = 64 -> 2x 16x16x32
            v16h a, b;
#pragma unroll
            for (int i = 0; i < 16; ++i) {
                // A layout: contiguous K = 16*hs + i (matches B)
                int ka = 16 * hs + i;
                a[i] = s_q[(mi * 16 + n) * 72 + kk * 32 + ka];
                // B layout: lanes0-15 rows K=0..15, lanes16-31 rows K=16..31, col = n
                b[i] = (_Float16)cur[(kk * 32 + 16 * hs + i) * 64 + si * 16 + n];
            }
            acc = __builtin_amdgcn_wmma_f32_16x16x32_f16(false, a, false, b,
                                                         (short)0, acc, false, false);
        }
        // epilogue: fuse *scale + prev, write f32 scores to HBM, f16 copy to LDS
#pragma unroll
        for (int c = 0; c < 8; ++c) {
            int m    = c + 8 * hs;                 // C/D layout: M = vgpr + 8*halfwave
            int grow = m0 + mi * 16 + m;
            int gcol = sc * 64 + si * 16 + n;
            size_t idx = ((size_t)(bh * QL + grow)) * SL + gcol;
            float sv = acc[c] * scl + prev[idx];
            out[S_OFF + idx] = sv;
            s_w[(mi * 16 + m) * WSTRIDE + gcol] = (_Float16)sv;
        }
    }
    __syncthreads();

    // ---- Stage 3: row softmax * mask ; f32 weights to HBM, f16 copy in LDS ----
    for (int j = 0; j < 4; ++j) {
        int r = wave * 4 + j;                      // each wave owns 4 rows
        _Float16* rowh = s_w + r * WSTRIDE;
        float mx = -3.402823466e38f;
        for (int c = lane; c < SL; c += 32) mx = fmaxf(mx, (float)rowh[c]);
#pragma unroll
        for (int off = 16; off; off >>= 1) mx = fmaxf(mx, __shfl_xor(mx, off, 32));
        float sum = 0.f;
        for (int c = lane; c < SL; c += 32) sum += __expf((float)rowh[c] - mx);
#pragma unroll
        for (int off = 16; off; off >>= 1) sum += __shfl_xor(sum, off, 32);
        float inv = 1.0f / sum;
        int grow = m0 + r;
        const float* mrow = mask + ((size_t)(h * QL + grow)) * SL;
        float* wrow = out + W_OFF + ((size_t)(bh * QL + grow)) * SL;
        for (int c = lane; c < SL; c += 32) {
            float wv = __expf((float)rowh[c] - mx) * inv * mrow[c];
            rowh[c] = (_Float16)wv;   // LDS copy feeds the PV matmul (native A-fragment type)
            wrow[c] = wv;             // returned attn_weights (f32)
        }
    }
    __syncthreads();

    // ---- Stage 4: output = weights @ v ; double-buffered async v stream ----
    const int ni = si;                             // 0..3 : output 16-col sub-tile
    v8f acc2 = {};
    issue_v(0, s_kv0);
    for (int kc = 0; kc < SL / 32; ++kc) {
        float* cur = (kc & 1) ? s_kv1 : s_kv0;
        float* nxt = (kc & 1) ? s_kv0 : s_kv1;
        wait_async0();
        __syncthreads();
        if (kc + 1 < SL / 32) issue_v(kc + 1, nxt);

        v16h a, b;
#pragma unroll
        for (int i = 0; i < 16; ++i) {
            int ka = 16 * hs + i;
            a[i] = s_w[(mi * 16 + n) * WSTRIDE + kc * 32 + ka];       // native f16, b128-coalesced
            b[i] = (_Float16)cur[(16 * hs + i) * 64 + ni * 16 + n];
        }
        acc2 = __builtin_amdgcn_wmma_f32_16x16x32_f16(false, a, false, b,
                                                      (short)0, acc2, false, false);
    }
#pragma unroll
    for (int c = 0; c < 8; ++c) {
        int m    = c + 8 * hs;
        int grow = m0 + mi * 16 + m;
        out[((size_t)(bh * QL + grow)) * HD + ni * 16 + n] = acc2[c];
    }
}

extern "C" void kernel_launch(void* const* d_in, const int* in_sizes, int n_in,
                              void* d_out, int out_size, void* d_ws, size_t ws_size,
                              hipStream_t stream) {
    const float* q     = (const float*)d_in[0];
    const float* k     = (const float*)d_in[1];
    const float* v     = (const float*)d_in[2];
    const float* prev  = (const float*)d_in[3];
    const float* mask  = (const float*)d_in[4];
    const float* scale = (const float*)d_in[5];
    float* out = (float*)d_out;

    dim3 grid(BH * (QL / MTILE));   // 2048 workgroups
    dim3 block(256);                // 8 wave32s
    attn_fused_kernel<<<grid, block, LDS_TOTAL, stream>>>(q, k, v, prev, mask, scale, out);
}
